// ParlerTTS_23252952940864
// MI455X (gfx1250) — matrix operand order repaired and checked by reading, not compiled, not run on hardware
//
#include <hip/hip_runtime.h>
#include <stdint.h>

// Model dims
#define B_    4
#define S_    576
#define SP_   64
#define SA_   512
#define SE_   64
#define H_    1024
#define NH_   16
#define HD_   64
#define FFN_  4096
#define L_    8
#define NCB_  9
#define VOC_  1088
#define NV_   (NCB_*VOC_)   // 9792

typedef __bf16 bf16_t;
typedef bf16_t v16bf __attribute__((ext_vector_type(16)));
typedef float  v8f   __attribute__((ext_vector_type(8)));

// pack two floats into two bf16 (RNE) in one u32; native cvt via (__bf16) cast
__device__ __forceinline__ unsigned int pkbf(float a, float b) {
  union { __bf16 h[2]; unsigned int u; } p;
  p.h[0] = (__bf16)a; p.h[1] = (__bf16)b;
  return p.u;
}
__device__ __forceinline__ unsigned short f2bf1(float a) {
  union { __bf16 h; unsigned short u; } p;
  p.h = (__bf16)a;
  return p.u;
}

struct GemmP {
  const float* A; long long lda, soA, siA;
  const float* B; long long ldb, soB, siB;
  float*       C; long long ldc, soC, siC;
  const float* bias;
  const float* R; long long ldr, soR, siR;
  int M; int K; int inner; float alpha; int flags; // bit0: B stored [N,K]; bit1: tanh-GELU
};

// 128x64 C tile per 256-thread block (8 wave32s, 4x2 wave grid).
// Each wave owns a 32x32 C tile = 4 f32 accumulators = 4 WMMAs per BK=32 step.
// A/B staged fp32 -> bf16 through LDS with float4 global loads.
__global__ __launch_bounds__(256) void gemm_wmma_bf16(GemmP p) {
  __shared__ __align__(16) unsigned int As[128][18];  // [m][k-pairs], 36 bf16/row padded
  __shared__ __align__(16) unsigned int Bs[64][18];   // [n][k-pairs]

  const int tid = threadIdx.x;
  const int z  = blockIdx.z;
  const int zo = z / p.inner, zi = z % p.inner;
  const float* A  = p.A + (long long)zo * p.soA + (long long)zi * p.siA;
  const float* Bm = p.B + (long long)zo * p.soB + (long long)zi * p.siB;
  float*       C  = p.C + (long long)zo * p.soC + (long long)zi * p.siC;
  const long long m0 = (long long)blockIdx.y * 128;
  const long long n0 = (long long)blockIdx.x * 64;
  const int M = p.M;

  const int w    = tid >> 5;
  const int lane = tid & 31;
  const int mw   = (w >> 1) * 32;   // wave m offset in tile
  const int nw   = (w & 1) * 32;    // wave n offset in tile
  const int hi   = lane >> 4;
  const int lo   = lane & 15;

  v8f acc[2][2];
  #pragma unroll
  for (int mi = 0; mi < 2; ++mi)
    #pragma unroll
    for (int ni = 0; ni < 2; ++ni)
      acc[mi][ni] = (v8f){0.f,0.f,0.f,0.f,0.f,0.f,0.f,0.f};

  for (int k0 = 0; k0 < p.K; k0 += 32) {
    __syncthreads();
    // ---- stage A tile 128x32: 1024 float4 loads, 4 per thread ----
    #pragma unroll
    for (int i = 0; i < 4; ++i) {
      int idx = tid + i * 256;
      int r = idx >> 3, c4 = idx & 7;                  // c = c4*4
      long long gr = m0 + r; if (gr > M - 1) gr = M - 1;  // clamp (epilogue guards writes)
      const float4 v = *(const float4*)&A[gr * p.lda + (k0 + c4 * 4)];
      As[r][c4 * 2]     = pkbf(v.x, v.y);
      As[r][c4 * 2 + 1] = pkbf(v.z, v.w);
    }
    // ---- stage B tile as [n][k] ----
    if (p.flags & 1) {   // B stored [N,K]: contiguous in k, vectorized
      #pragma unroll
      for (int i = 0; i < 2; ++i) {
        int idx = tid + i * 256;
        int r = idx >> 3, c4 = idx & 7;
        const float4 v = *(const float4*)&Bm[(long long)(n0 + r) * p.ldb + (k0 + c4 * 4)];
        Bs[r][c4 * 2]     = pkbf(v.x, v.y);
        Bs[r][c4 * 2 + 1] = pkbf(v.z, v.w);
      }
    } else {             // B stored [K,N] (weights): float4 along n, scatter to 4 rows
      unsigned short* Bh = (unsigned short*)&Bs[0][0];
      #pragma unroll
      for (int i = 0; i < 2; ++i) {
        int idx = tid + i * 256;
        int kk = idx >> 4, n4 = idx & 15;              // kk in 0..31, n = n4*4
        const float4 v = *(const float4*)&Bm[(long long)(k0 + kk) * p.ldb + (n0 + n4 * 4)];
        Bh[(n4 * 4 + 0) * 36 + kk] = f2bf1(v.x);
        Bh[(n4 * 4 + 1) * 36 + kk] = f2bf1(v.y);
        Bh[(n4 * 4 + 2) * 36 + kk] = f2bf1(v.z);
        Bh[(n4 * 4 + 3) * 36 + kk] = f2bf1(v.w);
      }
    }
    if (k0 + 32 < p.K) {  // prefetch a slice of the next A tile
      long long gr = m0 + (tid >> 1); if (gr > M - 1) gr = M - 1;
      __builtin_prefetch(&A[gr * p.lda + (k0 + 32) + ((tid & 1) << 4)], 0, 0);
    }
    __syncthreads();

    union Frag { unsigned int u[8]; v16bf v; };
    Frag fa[2], fb[2];
    #pragma unroll
    for (int mi = 0; mi < 2; ++mi) {
      // A 16x32 layout: lanes 0-15: K = 2i (+16 for i>=4); lanes 16-31: +8
      const unsigned int* rp = &As[mw + mi * 16 + lo][0];
      #pragma unroll
      for (int i = 0; i < 8; ++i) fa[mi].u[i] = rp[((i < 4) ? i : i + 4) + 4 * hi];
    }
    #pragma unroll
    for (int ni = 0; ni < 2; ++ni) {
      // B 32x16 layout: lanes 0-15: K = 2i (+16 for i>=4); lanes 16-31: +8
      const unsigned int* rp = &Bs[nw + ni * 16 + lo][0];
      #pragma unroll
      for (int i = 0; i < 8; ++i) fb[ni].u[i] = rp[((i < 4) ? i : i + 4) + 4 * hi];
    }
    #pragma unroll
    for (int mi = 0; mi < 2; ++mi)
      #pragma unroll
      for (int ni = 0; ni < 2; ++ni)
        acc[mi][ni] = __builtin_amdgcn_wmma_f32_16x16x32_bf16(
            false, fa[mi].v, false, fb[ni].v, (short)0, acc[mi][ni], false, false);
  }

  // Epilogue: C layout VGPR i -> M = i (lanes 0-15) / 8+i (lanes 16-31), N = lane%16
  const float* Rp = p.R ? (p.R + (long long)zo * p.soR + (long long)zi * p.siR) : nullptr;
  #pragma unroll
  for (int mi = 0; mi < 2; ++mi) {
    #pragma unroll
    for (int ni = 0; ni < 2; ++ni) {
      v8f a = acc[mi][ni];
      long long n = n0 + nw + ni * 16 + lo;
      float bi = p.bias ? p.bias[n] : 0.f;
      #pragma unroll
      for (int i = 0; i < 8; ++i) {
        long long m = m0 + mw + mi * 16 + i + 8 * hi;
        if (m < M) {
          float v = (a[i] + bi) * p.alpha;
          if (p.flags & 2) {
            float t = v;
            v = 0.5f * t * (1.f + tanhf(0.7978845608028654f * (t + 0.044715f * t * t * t)));
          }
          if (Rp) v += Rp[m * p.ldr + n];
          C[m * p.ldc + n] = v;
        }
      }
    }
  }
}

// One block per row of H_=1024: LayerNorm (f32 in, f32 out)
__global__ __launch_bounds__(256) void ln_kernel(const float* __restrict__ x,
                                                 const float* __restrict__ g,
                                                 const float* __restrict__ bb,
                                                 float* __restrict__ y) {
  __shared__ float red[256];
  int row = blockIdx.x, tid = threadIdx.x;
  const float* xr = x + (size_t)row * H_;
  float a[4]; float s = 0.f;
  #pragma unroll
  for (int i = 0; i < 4; ++i) { a[i] = xr[tid + i * 256]; s += a[i]; }
  red[tid] = s; __syncthreads();
  for (int o = 128; o > 0; o >>= 1) { if (tid < o) red[tid] += red[tid + o]; __syncthreads(); }
  float mean = red[0] * (1.f / H_); __syncthreads();
  float s2 = 0.f;
  #pragma unroll
  for (int i = 0; i < 4; ++i) { float d = a[i] - mean; s2 += d * d; }
  red[tid] = s2; __syncthreads();
  for (int o = 128; o > 0; o >>= 1) { if (tid < o) red[tid] += red[tid + o]; __syncthreads(); }
  float r = rsqrtf(red[0] * (1.f / H_) + 1e-5f);
  float* yr = y + (size_t)row * H_;
  #pragma unroll
  for (int i = 0; i < 4; ++i) { int c = tid + i * 256; yr[c] = (a[i] - mean) * r * g[c] + bb[c]; }
}

// In-place row softmax; optional causal mask (key index > query -> -1e9)
__global__ __launch_bounds__(256) void softmax_kernel(float* p, int len, int causal) {
  __shared__ float red[256];
  int row = blockIdx.x, tid = threadIdx.x;
  float* rp = p + (size_t)row * len;
  int q = row % S_;
  float mx = -3.4e38f;
  for (int c = tid; c < len; c += 256) {
    float v = rp[c]; if (causal && c > q) v = -1e9f;
    mx = fmaxf(mx, v);
  }
  red[tid] = mx; __syncthreads();
  for (int o = 128; o > 0; o >>= 1) { if (tid < o) red[tid] = fmaxf(red[tid], red[tid + o]); __syncthreads(); }
  mx = red[0]; __syncthreads();
  float sm = 0.f;
  for (int c = tid; c < len; c += 256) {
    float v = rp[c]; if (causal && c > q) v = -1e9f;
    sm += expf(v - mx);
  }
  red[tid] = sm; __syncthreads();
  for (int o = 128; o > 0; o >>= 1) { if (tid < o) red[tid] += red[tid + o]; __syncthreads(); }
  float inv = 1.f / red[0];
  for (int c = tid; c < len; c += 256) {
    float v = rp[c]; if (causal && c > q) v = -1e9f;
    rp[c] = expf(v - mx) * inv;
  }
}

// x[b,s,:] = (s<64 ? prompt : sum_cb embed_audio[cb][ids]) + pos_table[pos_ids]
__global__ __launch_bounds__(256) void embed_kernel(const int* __restrict__ ids,
                                                    const int* __restrict__ pos,
                                                    const float* __restrict__ tab,
                                                    const float* __restrict__ prompt,
                                                    const float* __restrict__ pt,
                                                    float* __restrict__ x) {
  int bs = blockIdx.x; int b = bs / S_, s = bs % S_;
  int tid = threadIdx.x;
  int pid = pos[b * S_ + s];
  float* xr = x + (size_t)bs * H_;
  for (int e = tid; e < H_; e += 256) {
    float v;
    if (s < SP_) {
      v = prompt[((size_t)b * SP_ + s) * H_ + e];
    } else {
      v = 0.f;
      int sa = s - SP_;
      #pragma unroll
      for (int cb = 0; cb < NCB_; ++cb) {
        int id = ids[((size_t)b * NCB_ + cb) * SA_ + sa];
        v += tab[((size_t)cb * (VOC_ + 1) + id) * H_ + e];
      }
    }
    v += pt[(size_t)pid * H_ + e];
    xr[e] = v;
  }
}

// [B,S,NCB,V] -> [B,NCB,S,V]
__global__ void reshape_kernel(const float* __restrict__ lg, float* __restrict__ out,
                               long long total) {
  long long i = (long long)blockIdx.x * blockDim.x + threadIdx.x;
  if (i >= total) return;
  long long v  = i % VOC_;
  long long t  = i / VOC_;
  long long s  = t % S_;
  long long t2 = t / S_;
  long long cb = t2 % NCB_;
  long long b  = t2 / NCB_;
  out[i] = lg[(b * S_ + s) * (long long)NV_ + cb * VOC_ + v];
}

static void launch_gemm(hipStream_t st,
    const float* A, long long lda, long long soA, long long siA,
    const float* B, long long ldb, long long soB, long long siB,
    float* C, long long ldc, long long soC, long long siC,
    const float* bias, const float* R, long long ldr, long long soR, long long siR,
    int M, int N, int K, int inner, int batches, float alpha, int flags) {
  GemmP p;
  p.A = A; p.lda = lda; p.soA = soA; p.siA = siA;
  p.B = B; p.ldb = ldb; p.soB = soB; p.siB = siB;
  p.C = C; p.ldc = ldc; p.soC = soC; p.siC = siC;
  p.bias = bias;
  p.R = R; p.ldr = ldr; p.soR = soR; p.siR = siR;
  p.M = M; p.K = K; p.inner = inner; p.alpha = alpha; p.flags = flags;
  dim3 g(N / 64, (M + 127) / 128, batches);
  hipLaunchKernelGGL(gemm_wmma_bf16, g, dim3(256), 0, st, p);
}

extern "C" void kernel_launch(void* const* d_in, const int* in_sizes, int n_in,
                              void* d_out, int out_size, void* d_ws, size_t ws_size,
                              hipStream_t stream) {
  (void)in_sizes; (void)n_in; (void)out_size; (void)ws_size;
  const int*   ids    = (const int*)d_in[0];
  const int*   pos    = (const int*)d_in[1];
  const float* tab    = (const float*)d_in[2];
  const float* prompt = (const float*)d_in[3];
  const float* enc    = (const float*)d_in[4];
  const float* pt     = (const float*)d_in[5];
  const float* ln_sa_g = (const float*)d_in[6];
  const float* ln_sa_b = (const float*)d_in[7];
  const float* wq_sa = (const float*)d_in[8];
  const float* bq_sa = (const float*)d_in[9];
  const float* wk_sa = (const float*)d_in[10];
  const float* wv_sa = (const float*)d_in[11];
  const float* bv_sa = (const float*)d_in[12];
  const float* wo_sa = (const float*)d_in[13];
  const float* bo_sa = (const float*)d_in[14];
  const float* ln_ca_g = (const float*)d_in[15];
  const float* ln_ca_b = (const float*)d_in[16];
  const float* wq_ca = (const float*)d_in[17];
  const float* bq_ca = (const float*)d_in[18];
  const float* wk_ca = (const float*)d_in[19];
  const float* wv_ca = (const float*)d_in[20];
  const float* bv_ca = (const float*)d_in[21];
  const float* wo_ca = (const float*)d_in[22];
  const float* bo_ca = (const float*)d_in[23];
  const float* ln_f_g = (const float*)d_in[24];
  const float* ln_f_b = (const float*)d_in[25];
  const float* w_fc1 = (const float*)d_in[26];
  const float* b_fc1 = (const float*)d_in[27];
  const float* w_fc2 = (const float*)d_in[28];
  const float* b_fc2 = (const float*)d_in[29];
  const float* lnorm_g = (const float*)d_in[30];
  const float* lnorm_b = (const float*)d_in[31];
  const float* lm_w   = (const float*)d_in[32];

  const long long NT = (long long)B_ * S_;           // 2304 token rows
  const long long HH = (long long)H_ * H_;
  float* ws = (float*)d_ws;
  float* X  = ws;                                    // residual stream f32 [NT,H]
  float* Hb = X  + NT * H_;                          // LN out / attn ctx
  float* Qb = Hb + NT * H_;
  float* Kb = Qb + NT * H_;
  float* Vb = Kb + NT * H_;
  float* SC = Vb + NT * H_;                          // scores [B,NH,S,S]
  float* FF = SC + (long long)B_ * NH_ * S_ * S_;    // ffn mid [NT,FFN]
  float* LG = SC;                                    // logits alias over SC(+FF)

  const float qscale = 0.125f;                       // HD^-0.5

  hipLaunchKernelGGL(embed_kernel, dim3(B_ * S_), dim3(256), 0, stream,
                     ids, pos, tab, prompt, pt, X);

  for (int l = 0; l < L_; ++l) {
    // ---- self attention ----
    hipLaunchKernelGGL(ln_kernel, dim3((unsigned)NT), dim3(256), 0, stream,
                       X, ln_sa_g + l * H_, ln_sa_b + l * H_, Hb);
    launch_gemm(stream, Hb, H_, 0, 0, wq_sa + l * HH, H_, 0, 0, Qb, H_, 0, 0,
                bq_sa + l * H_, nullptr, 0, 0, 0, (int)NT, H_, H_, 1, 1, qscale, 0);
    launch_gemm(stream, Hb, H_, 0, 0, wk_sa + l * HH, H_, 0, 0, Kb, H_, 0, 0,
                nullptr, nullptr, 0, 0, 0, (int)NT, H_, H_, 1, 1, 1.f, 0);
    launch_gemm(stream, Hb, H_, 0, 0, wv_sa + l * HH, H_, 0, 0, Vb, H_, 0, 0,
                bv_sa + l * H_, nullptr, 0, 0, 0, (int)NT, H_, H_, 1, 1, 1.f, 0);
    // scores[b,h] = Q_bh x K_bh^T  (B stored [N,K] -> flags bit0)
    launch_gemm(stream, Qb, H_, (long long)S_ * H_, HD_,
                Kb, H_, (long long)S_ * H_, HD_,
                SC, S_, (long long)NH_ * S_ * S_, (long long)S_ * S_,
                nullptr, nullptr, 0, 0, 0, S_, S_, HD_, NH_, B_ * NH_, 1.f, 1);
    hipLaunchKernelGGL(softmax_kernel, dim3(B_ * NH_ * S_), dim3(256), 0, stream, SC, S_, 1);
    // ctx[b,h] = P x V_bh -> packed into Hb [NT,H]
    launch_gemm(stream, SC, S_, (long long)NH_ * S_ * S_, (long long)S_ * S_,
                Vb, H_, (long long)S_ * H_, HD_,
                Hb, H_, (long long)S_ * H_, HD_,
                nullptr, nullptr, 0, 0, 0, S_, HD_, S_, NH_, B_ * NH_, 1.f, 0);
    launch_gemm(stream, Hb, H_, 0, 0, wo_sa + l * HH, H_, 0, 0, X, H_, 0, 0,
                bo_sa + l * H_, X, H_, 0, 0, (int)NT, H_, H_, 1, 1, 1.f, 0);

    // ---- cross attention ----
    hipLaunchKernelGGL(ln_kernel, dim3((unsigned)NT), dim3(256), 0, stream,
                       X, ln_ca_g + l * H_, ln_ca_b + l * H_, Hb);
    launch_gemm(stream, Hb, H_, 0, 0, wq_ca + l * HH, H_, 0, 0, Qb, H_, 0, 0,
                bq_ca + l * H_, nullptr, 0, 0, 0, (int)NT, H_, H_, 1, 1, qscale, 0);
    launch_gemm(stream, enc, H_, 0, 0, wk_ca + l * HH, H_, 0, 0, Kb, H_, 0, 0,
                nullptr, nullptr, 0, 0, 0, B_ * SE_, H_, H_, 1, 1, 1.f, 0);
    launch_gemm(stream, enc, H_, 0, 0, wv_ca + l * HH, H_, 0, 0, Vb, H_, 0, 0,
                bv_ca + l * H_, nullptr, 0, 0, 0, B_ * SE_, H_, H_, 1, 1, 1.f, 0);
    launch_gemm(stream, Qb, H_, (long long)S_ * H_, HD_,
                Kb, H_, (long long)SE_ * H_, HD_,
                SC, SE_, (long long)NH_ * S_ * SE_, (long long)S_ * SE_,
                nullptr, nullptr, 0, 0, 0, S_, SE_, HD_, NH_, B_ * NH_, 1.f, 1);
    hipLaunchKernelGGL(softmax_kernel, dim3(B_ * NH_ * S_), dim3(256), 0, stream, SC, SE_, 0);
    launch_gemm(stream, SC, SE_, (long long)NH_ * S_ * SE_, (long long)S_ * SE_,
                Vb, H_, (long long)SE_ * H_, HD_,
                Hb, H_, (long long)S_ * H_, HD_,
                nullptr, nullptr, 0, 0, 0, S_, HD_, SE_, NH_, B_ * NH_, 1.f, 0);
    launch_gemm(stream, Hb, H_, 0, 0, wo_ca + l * HH, H_, 0, 0, X, H_, 0, 0,
                bo_ca + l * H_, X, H_, 0, 0, (int)NT, H_, H_, 1, 1, 1.f, 0);

    // ---- FFN ----
    hipLaunchKernelGGL(ln_kernel, dim3((unsigned)NT), dim3(256), 0, stream,
                       X, ln_f_g + l * H_, ln_f_b + l * H_, Hb);
    launch_gemm(stream, Hb, H_, 0, 0, w_fc1 + (long long)l * H_ * FFN_, FFN_, 0, 0,
                FF, FFN_, 0, 0, b_fc1 + (long long)l * FFN_,
                nullptr, 0, 0, 0, (int)NT, FFN_, H_, 1, 1, 1.f, 2);
    launch_gemm(stream, FF, FFN_, 0, 0, w_fc2 + (long long)l * H_ * FFN_, H_, 0, 0,
                X, H_, 0, 0, b_fc2 + (long long)l * H_,
                X, H_, 0, 0, (int)NT, H_, FFN_, 1, 1, 1.f, 0);
  }

  // final LN + LM head + transpose to [B,NCB,S,V]
  hipLaunchKernelGGL(ln_kernel, dim3((unsigned)NT), dim3(256), 0, stream,
                     X, lnorm_g, lnorm_b, Hb);
  launch_gemm(stream, Hb, H_, 0, 0, lm_w, NV_, 0, 0, LG, NV_, 0, 0,
              nullptr, nullptr, 0, 0, 0, (int)NT, NV_, H_, 1, 1, 1.f, 0);
  long long total = (long long)B_ * NCB_ * S_ * VOC_;
  hipLaunchKernelGGL(reshape_kernel, dim3((unsigned)((total + 255) / 256)), dim3(256),
                     0, stream, LG, (float*)d_out, total);
}
